// L2P_76038101008832
// MI455X (gfx1250) — matrix operand order repaired and checked by reading, not compiled, not run on hardware
//
#include <hip/hip_runtime.h>

typedef __attribute__((ext_vector_type(16))) _Float16 v16h;
typedef __attribute__((ext_vector_type(8)))  float    v8f;

#define B_    128
#define NL_   12
#define C_    768
#define POOL  30
#define POOLP 32      // pool padded to 32 for WMMA N tiles
#define PLEN  5
#define EMB   768
#define TOPK  5

// ---------------------------------------------------------------------------
// Normalize x_query rows (fp32 -> f16). One block per row of 768.
// ---------------------------------------------------------------------------
__global__ void norm_q_kernel(const float* __restrict__ xq, _Float16* __restrict__ qh) {
    int row = blockIdx.x;               // row = b*NL_ + l
    int tid = threadIdx.x;              // 256 threads, 3 elems each
    const float* src = xq + (size_t)row * C_;
    float v0 = src[tid], v1 = src[tid + 256], v2 = src[tid + 512];
    __shared__ float red[256];
    red[tid] = v0 * v0 + v1 * v1 + v2 * v2;
    __syncthreads();
    for (int off = 128; off > 0; off >>= 1) {
        if (tid < off) red[tid] += red[tid + off];
        __syncthreads();
    }
    float inv = 1.0f / fmaxf(sqrtf(red[0]), 1e-12f);
    _Float16* dst = qh + (size_t)row * C_;
    dst[tid]       = (_Float16)(v0 * inv);
    dst[tid + 256] = (_Float16)(v1 * inv);
    dst[tid + 512] = (_Float16)(v2 * inv);
}

// ---------------------------------------------------------------------------
// Normalize e_k rows (fp32 -> f16), padded to POOLP rows per layer (pad = 0).
// ---------------------------------------------------------------------------
__global__ void norm_k_kernel(const float* __restrict__ ek, _Float16* __restrict__ nkh) {
    int p = blockIdx.x & (POOLP - 1);
    int l = blockIdx.x >> 5;
    int tid = threadIdx.x;
    _Float16* dst = nkh + ((size_t)l * POOLP + p) * C_;
    if (p >= POOL) {                    // zero pad rows so WMMA reads are clean
        dst[tid] = (_Float16)0.0f;
        dst[tid + 256] = (_Float16)0.0f;
        dst[tid + 512] = (_Float16)0.0f;
        return;
    }
    const float* src = ek + ((size_t)l * POOL + p) * C_;
    float v0 = src[tid], v1 = src[tid + 256], v2 = src[tid + 512];
    __shared__ float red[256];
    red[tid] = v0 * v0 + v1 * v1 + v2 * v2;
    __syncthreads();
    for (int off = 128; off > 0; off >>= 1) {
        if (tid < off) red[tid] += red[tid + off];
        __syncthreads();
    }
    float inv = 1.0f / fmaxf(sqrtf(red[0]), 1e-12f);
    dst[tid]       = (_Float16)(v0 * inv);
    dst[tid + 256] = (_Float16)(v1 * inv);
    dst[tid + 512] = (_Float16)(v2 * inv);
}

// ---------------------------------------------------------------------------
// cos_sim via v_wmma_f32_16x16x32_f16.
// Grid: NL_*8 blocks of 32 threads (one wave). Each wave computes a 16x32
// output tile: rows = 16 batch rows, cols = 32 (padded pool), K = 768.
// A lane layout (16-bit A 16x32): M = lane%16; reg v holds K pair
//   v<4: K = 2v + 8*half ; v>=4: K = 16 + 2(v-4) + 8*half   (half = lane/16)
// B lane layout (16-bit B 32x16): N = lane%16; reg v holds K pair same as A
// C layout: reg r -> row M = r + 8*half, col N = lane%16.
// ---------------------------------------------------------------------------
__global__ void cos_wmma_kernel(const _Float16* __restrict__ qh,
                                const _Float16* __restrict__ nkh,
                                float* __restrict__ cosb) {
    int l    = blockIdx.x >> 3;
    int mt   = blockIdx.x & 7;
    int lane = threadIdx.x;
    int half = lane >> 4;
    int r16  = lane & 15;
    int arow = mt * 16 + r16;           // batch row for A fragment

    const unsigned* qa = (const unsigned*)(qh + ((size_t)arow * NL_ + l) * C_);
    const unsigned* b0 = (const unsigned*)(nkh + ((size_t)l * POOLP + r16) * C_);
    const unsigned* b1 = (const unsigned*)(nkh + ((size_t)l * POOLP + 16 + r16) * C_);

    v8f acc0 = {};
    v8f acc1 = {};
    for (int k0 = 0; k0 < C_; k0 += 32) {
        union { unsigned u[8]; v16h h; } A, Bm0, Bm1;
        #pragma unroll
        for (int v = 0; v < 8; ++v) {
            int ka = (v < 4) ? (2 * v + 8 * half) : (16 + 2 * (v - 4) + 8 * half);
            int kb = ka;
            A.u[v]   = qa[(k0 + ka) >> 1];
            Bm0.u[v] = b0[(k0 + kb) >> 1];
            Bm1.u[v] = b1[(k0 + kb) >> 1];
        }
        acc0 = __builtin_amdgcn_wmma_f32_16x16x32_f16(
            false, A.h, false, Bm0.h, (short)0, acc0, false, false);
        acc1 = __builtin_amdgcn_wmma_f32_16x16x32_f16(
            false, A.h, false, Bm1.h, (short)0, acc1, false, false);
    }
    #pragma unroll
    for (int r = 0; r < 8; ++r) {
        int row = mt * 16 + r + 8 * half;
        float* o = cosb + ((size_t)row * NL_ + l) * POOLP;
        o[r16]      = acc0[r];
        o[16 + r16] = acc1[r];
    }
}

// ---------------------------------------------------------------------------
// Top-5 of 30 per (b,l); ascending-scan strict '>' argmax matches lax.top_k
// tie-breaking (lowest index wins). Loss partial = sum(1 - topval) because
// normalize(selected_K) == n_K[idx], so batched_sim == top-k values.
// ---------------------------------------------------------------------------
__global__ void topk_kernel(const float* __restrict__ cosb, int* __restrict__ kidx,
                            float* __restrict__ lpart) {
    int pair = blockIdx.x * blockDim.x + threadIdx.x;   // b*NL_ + l
    if (pair >= B_ * NL_) return;
    const float* v = cosb + (size_t)pair * POOLP;
    float vals[POOL];
    #pragma unroll
    for (int n = 0; n < POOL; ++n) vals[n] = v[n];
    int* ki = kidx + (size_t)pair * TOPK;
    float part = 0.0f;
    for (int t = 0; t < TOPK; ++t) {
        float best = -3.0e38f;
        int bi = 0;
        #pragma unroll
        for (int n = 0; n < POOL; ++n) {
            if (vals[n] > best) { best = vals[n]; bi = n; }
        }
        ki[t] = bi;
        part += 1.0f - best;
        vals[bi] = -3.0e38f;
    }
    lpart[pair] = part;
}

// ---------------------------------------------------------------------------
// Deterministic single-block reduction of 1536 loss partials -> loss.
// ---------------------------------------------------------------------------
__global__ void loss_reduce_kernel(const float* __restrict__ lpart,
                                   float* __restrict__ out_loss) {
    int tid = threadIdx.x;
    float s = 0.0f;
    for (int i = tid; i < B_ * NL_; i += 256) s += lpart[i];
    __shared__ float red[256];
    red[tid] = s;
    __syncthreads();
    for (int off = 128; off > 0; off >>= 1) {
        if (tid < off) red[tid] += red[tid + off];
        __syncthreads();
    }
    if (tid == 0) *out_loss = red[0] / (float)B_;
}

// ---------------------------------------------------------------------------
// Prompt gather: one block per (l,b,k) copies a contiguous 5x768-float block
// e_p[l, idx] -> out[l, b, k*5:(k+1)*5, :]. float4 => b128 loads/stores.
// ---------------------------------------------------------------------------
__global__ void gather_kernel(const float* __restrict__ ep, const int* __restrict__ kidx,
                              float* __restrict__ out) {
    int gid = blockIdx.x;               // l*B_*TOPK + b*TOPK + k
    int k = gid % TOPK;
    int b = (gid / TOPK) % B_;
    int l = gid / (TOPK * B_);
    int idx = kidx[((size_t)b * NL_ + l) * TOPK + k];
    const float4* src = (const float4*)(ep + ((size_t)(l * POOL + idx) * PLEN) * EMB);
    float4* dst = (float4*)(out + (((size_t)l * B_ + b) * (TOPK * PLEN) + k * PLEN) * EMB);
    const int n4 = PLEN * EMB / 4;      // 960 float4 = 15360 bytes
    for (int j = threadIdx.x; j < n4; j += 256) dst[j] = src[j];
}

// ---------------------------------------------------------------------------
extern "C" void kernel_launch(void* const* d_in, const int* in_sizes, int n_in,
                              void* d_out, int out_size, void* d_ws, size_t ws_size,
                              hipStream_t stream) {
    (void)in_sizes; (void)n_in; (void)out_size; (void)ws_size;
    const float* xq = (const float*)d_in[0];   // [128,12,768]
    const float* ep = (const float*)d_in[1];   // [12,30,5,768]
    const float* ek = (const float*)d_in[2];   // [12,30,768]
    float* out = (float*)d_out;                // [12,128,25,768] ++ [loss]

    // Workspace carve (all 256B-aligned region starts)
    char* w = (char*)d_ws;
    _Float16* qh   = (_Float16*)(w);                                  // 1536*768 f16 = 2359296 B
    _Float16* nkh  = (_Float16*)(w + 2359296);                        // 12*32*768 f16 = 589824 B
    float*    cosb = (float*)(w + 2359296 + 589824);                  // 128*12*32 f32 = 196608 B
    int*      kidx = (int*)(w + 2359296 + 589824 + 196608);           // 7680 i32 = 30720 B
    float*    lpt  = (float*)(w + 2359296 + 589824 + 196608 + 30720); // 1536 f32

    norm_q_kernel<<<B_ * NL_, 256, 0, stream>>>(xq, qh);
    norm_k_kernel<<<NL_ * POOLP, 256, 0, stream>>>(ek, nkh);
    cos_wmma_kernel<<<NL_ * 8, 32, 0, stream>>>(qh, nkh, cosb);
    topk_kernel<<<(B_ * NL_ + 255) / 256, 256, 0, stream>>>(cosb, kidx, lpt);
    loss_reduce_kernel<<<1, 256, 0, stream>>>(
        lpt, out + (size_t)NL_ * B_ * TOPK * PLEN * EMB);
    gather_kernel<<<NL_ * B_ * TOPK, 256, 0, stream>>>(ep, kidx, out);
}
